// LCGraphNet_63084479643693
// MI455X (gfx1250) — matrix operand order repaired and checked by reading, not compiled, not run on hardware
//
#include <hip/hip_runtime.h>
#include <math.h>

// ---------------------------------------------------------------------------
// CDNA5 / gfx1250 wave32 WMMA types
// ---------------------------------------------------------------------------
typedef __attribute__((ext_vector_type(16))) _Float16 v16h;
typedef __attribute__((ext_vector_type(8)))  _Float16 v8h;
typedef __attribute__((ext_vector_type(8)))  float    v8f;
typedef __attribute__((ext_vector_type(4)))  float    f32x4;

union Frag16 { v16h h; f32x4 q[2]; };
union Half8  { v8h  h; f32x4 q;    };

// A-matrix fragment (16x32 f16), loaded from a row-major [16, ld] tile.
// Per ISA 7.12.2: lane L holds row (L&15); lanes 0-15 hold K 0..7 (V0-3) and
// 16..23 (V4-7); lanes 16-31 hold K 8..15 and 24..31.
__device__ __forceinline__ v16h load_frag_a(const _Float16* __restrict__ base, int ld) {
  const int lane = threadIdx.x & 31;
  const _Float16* p = base + (size_t)(lane & 15) * ld + ((lane >> 4) << 3);
  Frag16 u;
  u.q[0] = *(const f32x4*)(p);        // K = k0 .. k0+7
  u.q[1] = *(const f32x4*)(p + 16);   // K = k0+16 .. k0+23
  return u.h;
}

// B-matrix fragment (32x16 f16). Column n of B is row n of a row-major
// [16, ld] source (keys / transposed weights): lane L holds column (L&15),
// K 0..7 and 16..23 (lanes 0-15) or 8..15 and 24..31 (lanes 16-31), same as A.
__device__ __forceinline__ v16h load_frag_b(const _Float16* __restrict__ base, int ld) {
  const int lane = threadIdx.x & 31;
  const _Float16* p = base + (size_t)(lane & 15) * ld + ((lane >> 4) << 3);
  Frag16 u;
  u.q[0] = *(const f32x4*)(p);
  u.q[1] = *(const f32x4*)(p + 16);
  return u.h;
}

// Replace the current worst entry of the register-resident top-8 and refresh
// the running max. Only executed on the rare "candidate wins" path.
__device__ __forceinline__ void top8_replace(float cd, int ci, float (&d)[8], int (&id)[8],
                                             float& mx) {
  float m = d[0]; int s = 0;
  #pragma unroll
  for (int i = 1; i < 8; ++i) { if (d[i] > m) { m = d[i]; s = i; } }
  #pragma unroll
  for (int i = 0; i < 8; ++i) { if (i == s) { d[i] = cd; id[i] = ci; } }
  m = d[0];
  #pragma unroll
  for (int i = 1; i < 8; ++i) m = fmaxf(m, d[i]);
  mx = m;
}

// ---------------------------------------------------------------------------
// Feature prep
// ---------------------------------------------------------------------------
__global__ void cast_pad_kernel(const float* __restrict__ X, _Float16* __restrict__ F,
                                int N, int d, int DP) {
  int i = blockIdx.x * blockDim.x + threadIdx.x;
  int total = N * DP;
  if (i >= total) return;
  int n = i / DP, c = i - n * DP;
  F[i] = (c < d) ? (_Float16)X[n * d + c] : (_Float16)0.0f;
}

__global__ void norms_kernel(const _Float16* __restrict__ F, float* __restrict__ nrm,
                             int N, int DP) {
  int i = blockIdx.x * blockDim.x + threadIdx.x;
  if (i >= N) return;
  float s = 0.0f;
  for (int c = 0; c < DP; ++c) { float v = (float)F[(size_t)i * DP + c]; s += v * v; }
  nrm[i] = s;
}

// W [K, NCOL] f32 row-major  ->  Wt [NCOL, K] f16 row-major (contiguous K per out col)
__global__ void wtrans_kernel(const float* __restrict__ W, _Float16* __restrict__ Wt,
                              int K, int NCOL) {
  int i = blockIdx.x * blockDim.x + threadIdx.x;
  if (i >= K * NCOL) return;
  int o = i / K, k = i - o * K;
  Wt[(size_t)o * K + k] = (_Float16)W[(size_t)k * NCOL + o];
}

// ---------------------------------------------------------------------------
// Fused kNN: WMMA Gram + streaming top-8. One wave (32 threads) per 16 queries.
// d2 is never materialized in global memory. The candidate scan batch-loads
// each lane's 16 candidates (64 contiguous bytes of LDS) as 4x b128, then uses
// a running-max fast reject (1 cmp) with a wave-branch around the replace.
// ---------------------------------------------------------------------------
template<int DP>
__global__ __launch_bounds__(32)
void knn_kernel(const _Float16* __restrict__ F, const float* __restrict__ nrm,
                int* __restrict__ knn_idx, int N) {
  constexpr int NC = DP / 32;             // k-chunks of 32
  const int lane = threadIdx.x & 31;
  const int half = lane >> 4;
  const int lc   = lane & 15;
  const int q0   = blockIdx.x * 16;

  // Query fragments held in registers for the whole sweep.
  v16h afr[NC];
  #pragma unroll
  for (int c = 0; c < NC; ++c)
    afr[c] = load_frag_a(F + (size_t)q0 * DP + c * 32, DP);

  // Row norms for the WMMA accumulator rows this lane produces.
  float rn[8];
  #pragma unroll
  for (int i = 0; i < 8; ++i) rn[i] = nrm[q0 + i + 8 * half];

  float bd[8]; int bi[8];
  #pragma unroll
  for (int i = 0; i < 8; ++i) { bd[i] = 3.0e38f; bi[i] = 0; }
  float curmax = 3.0e38f;

  __shared__ float tile[16 * 32];

  for (int j0 = 0; j0 < N; j0 += 32) {
    // Prefetch next key block toward L2/L0 (global_prefetch_b8).
    if (j0 + 32 < N)
      __builtin_prefetch(F + (size_t)(j0 + 32 + lc) * DP, 0, 1);

    #pragma unroll
    for (int t = 0; t < 2; ++t) {
      const int c0 = j0 + t * 16;
      v8f acc = {};
      #pragma unroll
      for (int c = 0; c < NC; ++c) {
        v16h bfr = load_frag_b(F + (size_t)c0 * DP + c * 32, DP);
        acc = __builtin_amdgcn_wmma_f32_16x16x32_f16(
            false, afr[c], false, bfr, (short)0, acc, false, false);
      }
      const float cn = nrm[c0 + lc];
      #pragma unroll
      for (int i = 0; i < 8; ++i)
        tile[(i + 8 * half) * 32 + t * 16 + lc] = rn[i] + cn - 2.0f * acc[i];
    }
    __syncthreads();

    // Update phase: lane owns row (lane&15), scans 16 columns of its half.
    // Batch the 16 candidates (64B, 64B-aligned) into registers first.
    f32x4 cq[4];
    const f32x4* tp = (const f32x4*)(tile + lc * 32 + half * 16);
    #pragma unroll
    for (int v = 0; v < 4; ++v) cq[v] = tp[v];
    #pragma unroll
    for (int c = 0; c < 16; ++c) {
      const float cand = cq[c >> 2][c & 3];
      if (cand < curmax) {                      // fast reject; rare taken path
        top8_replace(cand, j0 + half * 16 + c, bd, bi, curmax);
      }
    }
    __syncthreads();
  }

  // Merge the two lane halves (each row's candidates live in lane r and r+16).
  #pragma unroll
  for (int i = 0; i < 8; ++i) {
    float pd = __shfl(bd[i], lc + 16, 32);
    int   pi = __shfl(bi[i], lc + 16, 32);
    if (lane < 16 && pd < curmax) top8_replace(pd, pi, bd, bi, curmax);
  }
  if (lane < 16) {
    #pragma unroll
    for (int i = 0; i < 8; ++i) knn_idx[(size_t)(q0 + lc) * 8 + i] = bi[i];
  }
}

// ---------------------------------------------------------------------------
// Edge features: E[n*8+k] = [ x_n , x_{j} - x_n ]  (f16, width 2d), 16B vectors.
// ---------------------------------------------------------------------------
__global__ void edge_build_kernel(const _Float16* __restrict__ F, const int* __restrict__ knn_idx,
                                  _Float16* __restrict__ E, int N, int d, int DP) {
  const int dv = d >> 3;                      // 8-half vectors per row half
  int i = blockIdx.x * blockDim.x + threadIdx.x;
  int total = N * 8 * dv;
  if (i >= total) return;
  int r = i / dv, cv = i - r * dv;
  int c = cv << 3;
  int n = r >> 3;
  int j = knn_idx[r];
  Half8 xi, xj;
  xi.q = *(const f32x4*)(F + (size_t)n * DP + c);
  xj.q = *(const f32x4*)(F + (size_t)j * DP + c);
  Half8 dm; dm.h = xj.h - xi.h;
  *(f32x4*)(E + (size_t)r * (2 * d) + c)     = xi.q;
  *(f32x4*)(E + (size_t)r * (2 * d) + d + c) = dm.q;
}

// ---------------------------------------------------------------------------
// WMMA GEMM: Y[M, NCOL] (f32) = A[M, KD] (f16) @ Wt^T  (Wt is [NCOL, KD] f16)
// 8 waves per block; each wave computes a 32 x NCOL strip (2 row tiles) so
// every B (weight) fragment feeds two WMMAs.
// ---------------------------------------------------------------------------
template<int KD, int NCOL>
__global__ __launch_bounds__(256)
void gemm_kernel(const _Float16* __restrict__ A, const _Float16* __restrict__ Wt,
                 float* __restrict__ Y, int M) {
  constexpr int NT = NCOL / 16;
  constexpr int KC = KD / 32;
  const int lane = threadIdx.x & 31;
  const int wave = threadIdx.x >> 5;
  const int row0 = (blockIdx.x * 8 + wave) * 32;
  if (row0 >= M) return;

  v8f zero = {};
  v8f acc[2][NT];
  #pragma unroll
  for (int rt = 0; rt < 2; ++rt)
    #pragma unroll
    for (int t = 0; t < NT; ++t) acc[rt][t] = zero;

  #pragma unroll
  for (int kc = 0; kc < KC; ++kc) {
    v16h a0 = load_frag_a(A + (size_t)row0 * KD + kc * 32, KD);
    v16h a1 = load_frag_a(A + (size_t)(row0 + 16) * KD + kc * 32, KD);
    #pragma unroll
    for (int t = 0; t < NT; ++t) {
      v16h b = load_frag_b(Wt + (size_t)(t * 16) * KD + kc * 32, KD);
      acc[0][t] = __builtin_amdgcn_wmma_f32_16x16x32_f16(
          false, a0, false, b, (short)0, acc[0][t], false, false);
      acc[1][t] = __builtin_amdgcn_wmma_f32_16x16x32_f16(
          false, a1, false, b, (short)0, acc[1][t], false, false);
    }
  }

  const int half = lane >> 4, lc = lane & 15;
  #pragma unroll
  for (int rt = 0; rt < 2; ++rt) {
    #pragma unroll
    for (int t = 0; t < NT; ++t) {
      #pragma unroll
      for (int i = 0; i < 8; ++i)
        Y[(size_t)(row0 + rt * 16 + i + 8 * half) * NCOL + t * 16 + lc] = acc[rt][t][i];
    }
  }
}

// ---------------------------------------------------------------------------
// Training-mode BatchNorm statistics (biased variance), two-stage & coalesced.
// Stage 1: one block per 256-row chunk, row-contiguous reads.
// ---------------------------------------------------------------------------
__global__ __launch_bounds__(256)
void bn_stats1_kernel(const float* __restrict__ Y, float* __restrict__ part, int NCOL) {
  const int tpc = 256 / NCOL;                 // threads per channel (2 or 4)
  const int c   = threadIdx.x % NCOL;
  const int rg  = threadIdx.x / NCOL;
  const int row0 = blockIdx.x * 256;
  float s = 0.0f, s2 = 0.0f;
  for (int r = rg; r < 256; r += tpc) {
    float v = Y[(size_t)(row0 + r) * NCOL + c];
    s += v; s2 += v * v;
  }
  __shared__ float sh[512];
  sh[threadIdx.x] = s; sh[256 + threadIdx.x] = s2;
  __syncthreads();
  if (rg == 0) {
    for (int g = 1; g < tpc; ++g) { s += sh[g * NCOL + c]; s2 += sh[256 + g * NCOL + c]; }
    part[(size_t)blockIdx.x * 2 * NCOL + c]        = s;
    part[(size_t)blockIdx.x * 2 * NCOL + NCOL + c] = s2;
  }
}

// Stage 2: one small block folds the chunk partials into mean / rstd.
__global__ __launch_bounds__(128)
void bn_stats2_kernel(const float* __restrict__ part, float* __restrict__ mean,
                      float* __restrict__ rstd, int chunks, int NCOL, int M) {
  int c = threadIdx.x;
  if (c >= NCOL) return;
  float s = 0.0f, s2 = 0.0f;
  for (int k = 0; k < chunks; ++k) {
    s  += part[(size_t)k * 2 * NCOL + c];
    s2 += part[(size_t)k * 2 * NCOL + NCOL + c];
  }
  float m = s / (float)M;
  float var = s2 / (float)M - m * m;
  mean[c] = m;
  rstd[c] = rsqrtf(var + 1e-5f);
}

// BN + ReLU + cast to f16 activation matrix for the next WMMA GEMM.
__global__ void bn_act_kernel(const float* __restrict__ Y, const float* __restrict__ g,
                              const float* __restrict__ beta, const float* __restrict__ mean,
                              const float* __restrict__ rstd, _Float16* __restrict__ Aout,
                              int total, int NCOL) {
  int i = blockIdx.x * blockDim.x + threadIdx.x;
  if (i >= total) return;
  int c = i % NCOL;
  float sc = g[c] * rstd[c];
  float sh = beta[c] - mean[c] * sc;
  float v = fmaxf(Y[i] * sc + sh, 0.0f);
  Aout[i] = (_Float16)v;
}

// Final layer of a block: BN + ReLU + sum over K=8 neighbors -> node features.
__global__ void bn_agg_kernel(const float* __restrict__ Y, const float* __restrict__ g,
                              const float* __restrict__ beta, const float* __restrict__ mean,
                              const float* __restrict__ rstd, float* __restrict__ H,
                              _Float16* __restrict__ Fh, int N, int NCOL) {
  int i = blockIdx.x * blockDim.x + threadIdx.x;
  if (i >= N * NCOL) return;
  int n = i / NCOL, c = i - n * NCOL;
  float sc = g[c] * rstd[c];
  float sh = beta[c] - mean[c] * sc;
  float s = 0.0f;
  #pragma unroll
  for (int k = 0; k < 8; ++k) {
    float v = Y[(size_t)(n * 8 + k) * NCOL + c] * sc + sh;
    s += fmaxf(v, 0.0f);
  }
  H[i] = s;
  Fh[i] = (_Float16)s;
}

// ---------------------------------------------------------------------------
// Head: sigmoid( relu(H @ Wa + ba) @ Wb + bb ). One node per 128-thread block.
// ---------------------------------------------------------------------------
__global__ __launch_bounds__(128)
void head_kernel(const float* __restrict__ H, const float* __restrict__ Wa,
                 const float* __restrict__ ba, const float* __restrict__ Wb,
                 const float* __restrict__ bb, float* __restrict__ out, int N) {
  const int n = blockIdx.x;
  const int c = threadIdx.x;
  __shared__ float hrow[128];
  __shared__ float red[128];
  hrow[c] = H[(size_t)n * 128 + c];
  __syncthreads();
  float acc = ba[c];
  #pragma unroll 8
  for (int j = 0; j < 128; ++j) acc = fmaf(hrow[j], Wa[(size_t)j * 128 + c], acc);
  red[c] = fmaxf(acc, 0.0f) * Wb[c];
  __syncthreads();
  for (int o = 64; o > 0; o >>= 1) {
    if (c < o) red[c] += red[c + o];
    __syncthreads();
  }
  if (c == 0) out[n] = 1.0f / (1.0f + expf(-(red[0] + bb[0])));
}

// ---------------------------------------------------------------------------
// Driver
// ---------------------------------------------------------------------------
extern "C" void kernel_launch(void* const* d_in, const int* in_sizes, int n_in,
                              void* d_out, int out_size, void* d_ws, size_t ws_size,
                              hipStream_t stream) {
  (void)in_sizes; (void)n_in; (void)out_size; (void)ws_size;
  const int N = 16384;
  const int NK = N * 8;
  const int CHUNKS = NK / 256;             // 512 stat chunks
  const float* X = (const float*)d_in[0];
  auto P = [&](int i) { return (const float*)d_in[i]; };

  // --- carve workspace (all 256B aligned) ---
  char* w = (char*)d_ws;
  size_t off = 0;
  auto carve = [&](size_t bytes) -> void* {
    void* p = (void*)(w + off);
    off += (bytes + 255) & ~(size_t)255;
    return p;
  };
  _Float16* F1  = (_Float16*)carve((size_t)N * 32  * 2);  // padded input feats
  _Float16* F2  = (_Float16*)carve((size_t)N * 64  * 2);  // block1 out (f16)
  _Float16* F3  = (_Float16*)carve((size_t)N * 128 * 2);  // block2 out (f16)
  _Float16* F4  = (_Float16*)carve((size_t)N * 128 * 2);  // block3 out f16 (unused)
  float*    Hf  = (float*)   carve((size_t)N * 128 * 4);  // node feats f32
  float*    nrm = (float*)   carve((size_t)N * 4);
  int*      idx = (int*)     carve((size_t)N * 8 * 4);
  _Float16* Wt1 = (_Float16*)carve((size_t)256 * 128 * 2);
  _Float16* Wt2 = (_Float16*)carve((size_t)128 * 128 * 2);
  _Float16* Wt3 = (_Float16*)carve((size_t)128 * 128 * 2);
  float*    mean= (float*)   carve(128 * 4);
  float*    rstd= (float*)   carve(128 * 4);
  float*    part= (float*)   carve((size_t)CHUNKS * 2 * 128 * 4);
  _Float16* Eb  = (_Float16*)carve((size_t)NK * 256 * 2); // edges / reused act
  float*    Yb  = (float*)   carve((size_t)NK * 128 * 4); // layer output f32
  _Float16* Ab  = (_Float16*)carve((size_t)NK * 128 * 2); // activation f16

  auto bn_stats = [&](float* Ybuf, int ncol) {
    bn_stats1_kernel<<<CHUNKS, 256, 0, stream>>>(Ybuf, part, ncol);
    bn_stats2_kernel<<<1, 128, 0, stream>>>(part, mean, rstd, CHUNKS, ncol, NK);
  };

  // input: f32 -> padded f16 [N,32]
  cast_pad_kernel<<<(N * 32 + 255) / 256, 256, 0, stream>>>(X, F1, N, 16, 32);

  // ================= block 1: d=16 (DP=32), h=64, params base 1 =================
  norms_kernel<<<(N + 255) / 256, 256, 0, stream>>>(F1, nrm, N, 32);
  knn_kernel<32><<<N / 16, 32, 0, stream>>>(F1, nrm, idx, N);
  wtrans_kernel<<<(32 * 64 + 255) / 256, 256, 0, stream>>>(P(1), Wt1, 32, 64);
  wtrans_kernel<<<(64 * 64 + 255) / 256, 256, 0, stream>>>(P(5), Wt2, 64, 64);
  wtrans_kernel<<<(64 * 64 + 255) / 256, 256, 0, stream>>>(P(9), Wt3, 64, 64);
  edge_build_kernel<<<(NK * 2 + 255) / 256, 256, 0, stream>>>(F1, idx, Eb, N, 16, 32);
  gemm_kernel<32, 64><<<NK / 256, 256, 0, stream>>>(Eb, Wt1, Yb, NK);
  bn_stats(Yb, 64);
  bn_act_kernel<<<(NK * 64 + 255) / 256, 256, 0, stream>>>(Yb, P(3), P(4), mean, rstd, Ab, NK * 64, 64);
  gemm_kernel<64, 64><<<NK / 256, 256, 0, stream>>>(Ab, Wt2, Yb, NK);
  bn_stats(Yb, 64);
  bn_act_kernel<<<(NK * 64 + 255) / 256, 256, 0, stream>>>(Yb, P(7), P(8), mean, rstd, Eb, NK * 64, 64);
  gemm_kernel<64, 64><<<NK / 256, 256, 0, stream>>>(Eb, Wt3, Yb, NK);
  bn_stats(Yb, 64);
  bn_agg_kernel<<<(N * 64 + 255) / 256, 256, 0, stream>>>(Yb, P(11), P(12), mean, rstd, Hf, F2, N, 64);

  // ================= block 2: d=64, h=128, params base 13 =================
  norms_kernel<<<(N + 255) / 256, 256, 0, stream>>>(F2, nrm, N, 64);
  knn_kernel<64><<<N / 16, 32, 0, stream>>>(F2, nrm, idx, N);
  wtrans_kernel<<<(128 * 128 + 255) / 256, 256, 0, stream>>>(P(13), Wt1, 128, 128);
  wtrans_kernel<<<(128 * 128 + 255) / 256, 256, 0, stream>>>(P(17), Wt2, 128, 128);
  wtrans_kernel<<<(128 * 128 + 255) / 256, 256, 0, stream>>>(P(21), Wt3, 128, 128);
  edge_build_kernel<<<(NK * 8 + 255) / 256, 256, 0, stream>>>(F2, idx, Eb, N, 64, 64);
  gemm_kernel<128, 128><<<NK / 256, 256, 0, stream>>>(Eb, Wt1, Yb, NK);
  bn_stats(Yb, 128);
  bn_act_kernel<<<(NK * 128 + 255) / 256, 256, 0, stream>>>(Yb, P(15), P(16), mean, rstd, Ab, NK * 128, 128);
  gemm_kernel<128, 128><<<NK / 256, 256, 0, stream>>>(Ab, Wt2, Yb, NK);
  bn_stats(Yb, 128);
  bn_act_kernel<<<(NK * 128 + 255) / 256, 256, 0, stream>>>(Yb, P(19), P(20), mean, rstd, Eb, NK * 128, 128);
  gemm_kernel<128, 128><<<NK / 256, 256, 0, stream>>>(Eb, Wt3, Yb, NK);
  bn_stats(Yb, 128);
  bn_agg_kernel<<<(N * 128 + 255) / 256, 256, 0, stream>>>(Yb, P(23), P(24), mean, rstd, Hf, F3, N, 128);

  // ================= block 3: d=128, h=128, params base 25 =================
  norms_kernel<<<(N + 255) / 256, 256, 0, stream>>>(F3, nrm, N, 128);
  knn_kernel<128><<<N / 16, 32, 0, stream>>>(F3, nrm, idx, N);
  wtrans_kernel<<<(256 * 128 + 255) / 256, 256, 0, stream>>>(P(25), Wt1, 256, 128);
  wtrans_kernel<<<(128 * 128 + 255) / 256, 256, 0, stream>>>(P(29), Wt2, 128, 128);
  wtrans_kernel<<<(128 * 128 + 255) / 256, 256, 0, stream>>>(P(33), Wt3, 128, 128);
  edge_build_kernel<<<(NK * 16 + 255) / 256, 256, 0, stream>>>(F3, idx, Eb, N, 128, 128);
  gemm_kernel<256, 128><<<NK / 256, 256, 0, stream>>>(Eb, Wt1, Yb, NK);
  bn_stats(Yb, 128);
  bn_act_kernel<<<(NK * 128 + 255) / 256, 256, 0, stream>>>(Yb, P(27), P(28), mean, rstd, Ab, NK * 128, 128);
  gemm_kernel<128, 128><<<NK / 256, 256, 0, stream>>>(Ab, Wt2, Yb, NK);
  bn_stats(Yb, 128);
  bn_act_kernel<<<(NK * 128 + 255) / 256, 256, 0, stream>>>(Yb, P(31), P(32), mean, rstd, Eb, NK * 128, 128);
  gemm_kernel<128, 128><<<NK / 256, 256, 0, stream>>>(Eb, Wt3, Yb, NK);
  bn_stats(Yb, 128);
  bn_agg_kernel<<<(N * 128 + 255) / 256, 256, 0, stream>>>(Yb, P(35), P(36), mean, rstd, Hf, F4, N, 128);

  // ================= head =================
  head_kernel<<<N, 128, 0, stream>>>(Hf, P(37), P(38), P(39), P(40), (float*)d_out, N);
}
